// TransformerBlock_25125558681778
// MI455X (gfx1250) — matrix operand order repaired and checked by reading, not compiled, not run on hardware
//
#include <hip/hip_runtime.h>
#include <hip/hip_bf16.h>
#include <math.h>

// ---------------------------------------------------------------------------
// Types for CDNA5 WMMA (wave32): D(f32 16x16) = A(bf16 16x32) * B(bf16 32x16) + C
// ---------------------------------------------------------------------------
typedef __bf16 bf16;
typedef __attribute__((ext_vector_type(16))) __bf16 v16bf;
typedef __attribute__((ext_vector_type(8)))  __bf16 v8bf;
typedef __attribute__((ext_vector_type(8)))  float  v8f;
typedef __attribute__((ext_vector_type(4)))  int    v4i;

union Frag16 { v16bf v; v8bf h[2]; };

#if defined(__has_builtin)
#if __has_builtin(__builtin_amdgcn_global_load_async_to_lds_b128)
#define HAVE_ASYNC_LDS 1
#endif
#endif

static __device__ __forceinline__ v8f wmma_bf16(v16bf a, v16bf b, v8f c) {
  // (neg_a, A, neg_b, B, c_mod, C, reuse_a, reuse_b)
  return __builtin_amdgcn_wmma_f32_16x16x32_bf16(false, a, false, b, (short)0, c,
                                                 false, false);
}

// 16-byte global -> LDS copy, async (ASYNCcnt) when available.
static __device__ __forceinline__ void copy16_g2l(const bf16* g, bf16* l) {
#ifdef HAVE_ASYNC_LDS
  __builtin_amdgcn_global_load_async_to_lds_b128(
      (__attribute__((address_space(1))) v4i*)(v4i*)(void*)g,
      (__attribute__((address_space(3))) v4i*)(v4i*)(void*)l, 0, 0);
#else
  *(v8bf*)l = *(const v8bf*)g;
#endif
}

static __device__ __forceinline__ void async_wait0() {
#ifdef HAVE_ASYNC_LDS
#if __has_builtin(__builtin_amdgcn_s_wait_asynccnt)
  __builtin_amdgcn_s_wait_asynccnt(0);
#else
  asm volatile("s_wait_asynccnt 0" ::: "memory");
#endif
#endif
}

static __device__ __forceinline__ float gelu_tanh(float x) {
  const float c = 0.7978845608028654f; // sqrt(2/pi)
  return 0.5f * x * (1.0f + tanhf(c * (x + 0.044715f * x * x * x)));
}

// ---------------------------------------------------------------------------
// f32 -> bf16 elementwise convert (weights)
// ---------------------------------------------------------------------------
__global__ void cvt_f32_bf16_kernel(const float* __restrict__ in,
                                    bf16* __restrict__ out, int n) {
  int i = blockIdx.x * blockDim.x + threadIdx.x;
  int stride = gridDim.x * blockDim.x;
  for (; i < n; i += stride) out[i] = (bf16)in[i];
}

// ---------------------------------------------------------------------------
// LayerNorm over rows of 768 (torch unbiased var: /(N-1)), bf16 output
// ---------------------------------------------------------------------------
__global__ void __launch_bounds__(256)
ln_kernel(const float* __restrict__ x, const float* __restrict__ w,
          const float* __restrict__ b, bf16* __restrict__ out) {
  const int C = 768;
  int row = blockIdx.x;
  int tid = threadIdx.x;
  const float* xr = x + (size_t)row * C;

  float v[3];
#pragma unroll
  for (int i = 0; i < 3; ++i) v[i] = xr[tid + i * 256];

  __shared__ float red[256];
  float s = v[0] + v[1] + v[2];
  red[tid] = s;
  __syncthreads();
  for (int st = 128; st > 0; st >>= 1) {
    if (tid < st) red[tid] += red[tid + st];
    __syncthreads();
  }
  float mean = red[0] * (1.0f / 768.0f);
  __syncthreads();

  float sq = 0.0f;
#pragma unroll
  for (int i = 0; i < 3; ++i) { float d = v[i] - mean; sq += d * d; }
  red[tid] = sq;
  __syncthreads();
  for (int st = 128; st > 0; st >>= 1) {
    if (tid < st) red[tid] += red[tid + st];
    __syncthreads();
  }
  float var = red[0] * (1.0f / 767.0f);
  float rstd = rsqrtf(var + 1e-5f);

#pragma unroll
  for (int i = 0; i < 3; ++i) {
    int idx = tid + i * 256;
    out[(size_t)row * C + idx] = (bf16)((v[i] - mean) * rstd * w[idx] + b[idx]);
  }
}

// ---------------------------------------------------------------------------
// WMMA GEMM: out[M,N] = act[M,K](bf16) @ W[K,N](bf16) + bias, epilogues:
//   EPI==0 : bf16 out
//   EPI==1 : tanh-GELU, bf16 out
//   EPI==2 : + resid(f32), f32 out
// BM=BN=128, BK=32, 256 threads (8 waves, wave tile 64x32 = 4x2 WMMA tiles)
// A tile staged via async global->LDS; B tile transposed on store so that a
// B fragment is one contiguous 32B LDS read in the same k = 16h + s order as A.
// ---------------------------------------------------------------------------
template <int EPI>
__global__ void __launch_bounds__(256)
gemm_bf16_kernel(const bf16* __restrict__ A, const bf16* __restrict__ B,
                 const float* __restrict__ bias, const float* __restrict__ resid,
                 bf16* __restrict__ outb, float* __restrict__ outf,
                 int M, int N, int K) {
  __shared__ __align__(32) bf16 Alds[128][32]; // [m][k]
  __shared__ __align__(32) bf16 Blds[128][32]; // [n][k]  (transposed on store)

  const int tid = threadIdx.x;
  const int lane = tid & 31;
  const int w = tid >> 5;
  const int ln16 = lane & 15;
  const int hi = lane >> 4;
  const int koff16 = hi * 16;
  const int koff8 = hi * 8;
  const int wrow = w >> 2;   // 0..1
  const int wcol = w & 3;    // 0..3
  const int m0 = blockIdx.y * 128;
  const int n0 = blockIdx.x * 128;

  v8f acc[4][2];
#pragma unroll
  for (int mi = 0; mi < 4; ++mi)
#pragma unroll
    for (int ni = 0; ni < 2; ++ni)
#pragma unroll
      for (int e = 0; e < 8; ++e) acc[mi][ni][e] = 0.0f;

  const int arow = tid >> 1;        // 0..127
  const int ak = (tid & 1) * 16;    // 0,16
  const int bk = tid >> 3;          // 0..31
  const int bn = (tid & 7) * 16;    // 0..112

  for (int k0 = 0; k0 < K; k0 += 32) {
    __syncthreads();
    // stage A tile 128x32 (async global -> LDS, ASYNCcnt tracked)
    const bf16* ag = A + (size_t)(m0 + arow) * K + k0 + ak;
    copy16_g2l(ag,     &Alds[arow][ak]);
    copy16_g2l(ag + 8, &Alds[arow][ak + 8]);
    // stage B tile 32x128, transposed into [n][k]
    const bf16* bg = B + (size_t)(k0 + bk) * N + n0 + bn;
#pragma unroll
    for (int i = 0; i < 16; ++i) Blds[bn + i][bk] = bg[i];

    if (k0 + 32 < K && lane == 0) {
      __builtin_prefetch(A + (size_t)(m0 + w * 16) * K + k0 + 32, 0, 1);
      __builtin_prefetch(B + (size_t)(k0 + 32 + w * 4) * N + n0, 0, 1);
    }
    async_wait0();
    __syncthreads();

    Frag16 af[4];
    v16bf bfr[2];
#pragma unroll
    for (int mi = 0; mi < 4; ++mi) {
      int row = wrow * 64 + mi * 16 + ln16;
      af[mi].h[0] = *(const v8bf*)&Alds[row][koff16];
      af[mi].h[1] = *(const v8bf*)&Alds[row][koff16 + 8];
    }
#pragma unroll
    for (int ni = 0; ni < 2; ++ni) {
      int col = wcol * 32 + ni * 16 + ln16;
      bfr[ni] = *(const v16bf*)&Blds[col][koff16];
    }
#pragma unroll
    for (int mi = 0; mi < 4; ++mi)
#pragma unroll
      for (int ni = 0; ni < 2; ++ni)
        acc[mi][ni] = wmma_bf16(af[mi].v, bfr[ni], acc[mi][ni]);
  }

  // epilogue
#pragma unroll
  for (int mi = 0; mi < 4; ++mi)
#pragma unroll
    for (int ni = 0; ni < 2; ++ni)
#pragma unroll
      for (int r = 0; r < 8; ++r) {
        int row = m0 + wrow * 64 + mi * 16 + r + 8 * hi;
        int col = n0 + wcol * 32 + ni * 16 + ln16;
        float val = acc[mi][ni][r] + bias[col];
        if (EPI == 1) val = gelu_tanh(val);
        if (EPI == 2) {
          val += resid[(size_t)row * N + col];
          outf[(size_t)row * N + col] = val;
        } else {
          outb[(size_t)row * N + col] = (bf16)val;
        }
      }
}

// ---------------------------------------------------------------------------
// Causal flash attention. qkv[B*T, 2304] bf16 (q|k|v), ctx[B*T, 768] bf16.
// Block = 4 waves, each wave owns 16 q rows; key blocks of 32.
// K tile staged async; V transposed on store for contiguous B fragments.
// ---------------------------------------------------------------------------
__global__ void __launch_bounds__(128)
attn_kernel(const bf16* __restrict__ qkv, bf16* __restrict__ ctx) {
  const int T = 2048, C3 = 2304;
  const int qt0 = blockIdx.x * 64;  // q tile start within sequence
  const int h = blockIdx.y;
  const int b = blockIdx.z;
  const int tid = threadIdx.x;
  const int w = tid >> 5;
  const int lane = tid & 31;
  const int ln16 = lane & 15;
  const int hi = lane >> 4;
  const int koff16 = hi * 16;
  const int koff8 = hi * 8;

  __shared__ __align__(32) bf16 Klds[32][64];     // [key][d]
  __shared__ __align__(32) bf16 VldsT[64][32];    // [d][key]
  __shared__ __align__(32) bf16 Plds[4][16][32];  // per-wave P staging

  // Load this wave's Q tile (16 rows x 64) as two A fragments (d chunks of 32)
  const int qrow0 = qt0 + w * 16;
  Frag16 qa[2];
  {
    const bf16* qb = qkv + (size_t)(b * T + qrow0 + ln16) * C3 + h * 64;
#pragma unroll
    for (int dc = 0; dc < 2; ++dc) {
      qa[dc].h[0] = *(const v8bf*)(qb + dc * 32 + koff16);
      qa[dc].h[1] = *(const v8bf*)(qb + dc * 32 + koff16 + 8);
    }
  }

  float mrow[8], lrow[8];
  v8f O[4];
#pragma unroll
  for (int r = 0; r < 8; ++r) { mrow[r] = -1e30f; lrow[r] = 0.0f; }
#pragma unroll
  for (int t = 0; t < 4; ++t)
#pragma unroll
    for (int r = 0; r < 8; ++r) O[t][r] = 0.0f;

  const int nkb = qt0 / 32 + 2;  // key blocks covering keys <= qt0+63
  for (int kb = 0; kb < nkb; ++kb) {
    __syncthreads();
    {  // cooperative K/V staging (128 threads)
      int key = tid >> 2;             // 0..31
      int dpart = (tid & 3) * 16;     // 0,16,32,48
      const bf16* kp = qkv + (size_t)(b * T + kb * 32 + key) * C3 + 768 + h * 64 + dpart;
      copy16_g2l(kp,     &Klds[key][dpart]);
      copy16_g2l(kp + 8, &Klds[key][dpart + 8]);
      const bf16* vp = qkv + (size_t)(b * T + kb * 32 + key) * C3 + 1536 + h * 64 + dpart;
#pragma unroll
      for (int i = 0; i < 16; ++i) VldsT[dpart + i][key] = vp[i];
    }
    async_wait0();
    __syncthreads();

    bool active = (kb * 32) <= (qrow0 + 15);  // wave-uniform
    if (active) {
      // scores: two 16-key tiles, each = sum over 2 d-chunks of WMMA
      v8f s[2];
#pragma unroll
      for (int kt = 0; kt < 2; ++kt) {
#pragma unroll
        for (int e = 0; e < 8; ++e) s[kt][e] = 0.0f;
#pragma unroll
        for (int dc = 0; dc < 2; ++dc) {
          v16bf bk = *(const v16bf*)&Klds[kt * 16 + ln16][dc * 32 + koff16];
          s[kt] = wmma_bf16(qa[dc].v, bk, s[kt]);
        }
      }
      // scale + causal mask (D layout: row = r + 8*hi, col = ln16)
#pragma unroll
      for (int kt = 0; kt < 2; ++kt)
#pragma unroll
        for (int r = 0; r < 8; ++r) {
          int key = kb * 32 + kt * 16 + ln16;
          int q = qrow0 + r + 8 * hi;
          float sv = s[kt][r] * 0.125f;  // 1/sqrt(64)
          s[kt][r] = (key <= q) ? sv : -1e30f;
        }
      // online softmax update per row, stage P to LDS
#pragma unroll
      for (int r = 0; r < 8; ++r) {
        float tm = fmaxf(s[0][r], s[1][r]);
        for (int msk = 1; msk < 16; msk <<= 1) tm = fmaxf(tm, __shfl_xor(tm, msk, 32));
        float nm = fmaxf(mrow[r], tm);
        float sc = __expf(mrow[r] - nm);
        float p0 = __expf(s[0][r] - nm);
        float p1 = __expf(s[1][r] - nm);
        float rs = p0 + p1;
        for (int msk = 1; msk < 16; msk <<= 1) rs += __shfl_xor(rs, msk, 32);
        lrow[r] = lrow[r] * sc + rs;
        mrow[r] = nm;
#pragma unroll
        for (int t = 0; t < 4; ++t) O[t][r] *= sc;
        Plds[w][r + 8 * hi][ln16]      = (bf16)p0;
        Plds[w][r + 8 * hi][16 + ln16] = (bf16)p1;
      }
      // P(16x32) @ V(32x64): 4 WMMAs
      Frag16 pf;
      pf.h[0] = *(const v8bf*)&Plds[w][ln16][koff16];
      pf.h[1] = *(const v8bf*)&Plds[w][ln16][koff16 + 8];
#pragma unroll
      for (int t = 0; t < 4; ++t) {
        v16bf vf = *(const v16bf*)&VldsT[t * 16 + ln16][koff16];
        O[t] = wmma_bf16(pf.v, vf, O[t]);
      }
    }
  }

  // normalize and write context (bf16)
#pragma unroll
  for (int t = 0; t < 4; ++t)
#pragma unroll
    for (int r = 0; r < 8; ++r) {
      int tok = b * T + qrow0 + r + 8 * hi;
      int col = h * 64 + t * 16 + ln16;
      ctx[(size_t)tok * 768 + col] = (bf16)(O[t][r] / lrow[r]);
    }
}

// ---------------------------------------------------------------------------
// Orchestration
// ---------------------------------------------------------------------------
extern "C" void kernel_launch(void* const* d_in, const int* in_sizes, int n_in,
                              void* d_out, int out_size, void* d_ws, size_t ws_size,
                              hipStream_t stream) {
  (void)in_sizes; (void)n_in; (void)out_size; (void)ws_size;
  const int B = 4, T = 2048, C = 768;
  const int M = B * T;  // 8192

  const float* x        = (const float*)d_in[0];
  const float* ln1_w    = (const float*)d_in[1];
  const float* ln1_b    = (const float*)d_in[2];
  const float* c_attn_w = (const float*)d_in[3];
  const float* c_attn_b = (const float*)d_in[4];
  const float* c_proj_w = (const float*)d_in[5];
  const float* c_proj_b = (const float*)d_in[6];
  const float* ln2_w    = (const float*)d_in[7];
  const float* ln2_b    = (const float*)d_in[8];
  const float* fc_w     = (const float*)d_in[9];
  const float* fc_b     = (const float*)d_in[10];
  const float* mproj_w  = (const float*)d_in[11];
  const float* mproj_b  = (const float*)d_in[12];
  float* out = (float*)d_out;

  // workspace layout (bytes, all 256-aligned)
  char* ws = (char*)d_ws;
  constexpr size_t OFF_WQKV  = 0;                                   // 768*2304 bf16
  constexpr size_t OFF_WPROJ = OFF_WQKV  + (size_t)768 * 2304 * 2;  // 768*768  bf16
  constexpr size_t OFF_WFC   = OFF_WPROJ + (size_t)768 * 768 * 2;   // 768*3072 bf16
  constexpr size_t OFF_WMPJ  = OFF_WFC   + (size_t)768 * 3072 * 2;  // 3072*768 bf16
  constexpr size_t OFF_LN    = OFF_WMPJ  + (size_t)3072 * 768 * 2;  // 8192*768 bf16 (ln1, reused ln2)
  constexpr size_t OFF_QKV   = OFF_LN    + (size_t)8192 * 768 * 2;  // 8192*2304 bf16
  constexpr size_t OFF_CTX   = OFF_QKV   + (size_t)8192 * 2304 * 2; // 8192*768 bf16
  constexpr size_t OFF_X2    = OFF_CTX   + (size_t)8192 * 768 * 2;  // 8192*768 f32
  constexpr size_t OFF_FC    = OFF_X2    + (size_t)8192 * 768 * 4;  // 8192*3072 bf16

  bf16*  wqkv  = (bf16*)(ws + OFF_WQKV);
  bf16*  wproj = (bf16*)(ws + OFF_WPROJ);
  bf16*  wfc   = (bf16*)(ws + OFF_WFC);
  bf16*  wmpj  = (bf16*)(ws + OFF_WMPJ);
  bf16*  lnbuf = (bf16*)(ws + OFF_LN);
  bf16*  qkvb  = (bf16*)(ws + OFF_QKV);
  bf16*  ctxb  = (bf16*)(ws + OFF_CTX);
  float* x2    = (float*)(ws + OFF_X2);
  bf16*  fcbuf = (bf16*)(ws + OFF_FC);

  // 1. weight conversion to bf16
  cvt_f32_bf16_kernel<<<512, 256, 0, stream>>>(c_attn_w, wqkv, 768 * 2304);
  cvt_f32_bf16_kernel<<<512, 256, 0, stream>>>(c_proj_w, wproj, 768 * 768);
  cvt_f32_bf16_kernel<<<512, 256, 0, stream>>>(fc_w,     wfc,   768 * 3072);
  cvt_f32_bf16_kernel<<<512, 256, 0, stream>>>(mproj_w,  wmpj,  3072 * 768);

  // 2. ln1(x) -> bf16
  ln_kernel<<<M, 256, 0, stream>>>(x, ln1_w, ln1_b, lnbuf);

  // 3. qkv = ln1 @ Wqkv + b   [8192 x 2304]
  gemm_bf16_kernel<0><<<dim3(2304 / 128, M / 128), 256, 0, stream>>>(
      lnbuf, wqkv, c_attn_b, nullptr, qkvb, nullptr, M, 2304, C);

  // 4. causal flash attention -> ctx bf16 [8192 x 768]
  attn_kernel<<<dim3(T / 64, 12, B), 128, 0, stream>>>(qkvb, ctxb);

  // 5. x2 = x + ctx @ Wproj + b   (f32)
  gemm_bf16_kernel<2><<<dim3(768 / 128, M / 128), 256, 0, stream>>>(
      ctxb, wproj, c_proj_b, x, nullptr, x2, M, 768, C);

  // 6. ln2(x2) -> bf16 (reuse lnbuf)
  ln_kernel<<<M, 256, 0, stream>>>(x2, ln2_w, ln2_b, lnbuf);

  // 7. fc = gelu(ln2 @ Wfc + b) -> bf16 [8192 x 3072]
  gemm_bf16_kernel<1><<<dim3(3072 / 128, M / 128), 256, 0, stream>>>(
      lnbuf, wfc, fc_b, nullptr, fcbuf, nullptr, M, 3072, C);

  // 8. out = x2 + fc @ Wmproj + b   (f32)
  gemm_bf16_kernel<2><<<dim3(768 / 128, M / 128), 256, 0, stream>>>(
      fcbuf, wmpj, mproj_b, x2, nullptr, out, M, 768, 3072);
}
